// Masked_Attention_9680856286007
// MI455X (gfx1250) — matrix operand order repaired and checked by reading, not compiled, not run on hardware
//
#include <hip/hip_runtime.h>
#include <hip/hip_bf16.h>

// ---------------------------------------------------------------------------
// Causal MHA forward for MI455X (gfx1250, wave32, WMMA + TDM).
//   B=2, S=2048, E=1024, H=16, D=64.
// Matmuls -> V_WMMA_F32_16X16X32_F16 (f16 in, f32 accumulate).
// d_ws (f16): Q[b,h,s,d] | K[b,h,s,d] | V^T[b,h,d,s] | AttnOut[4096,1024]
//   = 4 * 8 MB = 32 MB.
// GEMMs: 128x128 tiles, BK=32, double-buffered LDS, software pipelined,
//        all fragment traffic as 16B vector ops.
// Attention: flash-style, block = 64 q rows x one (b,h); 64-key K/V tiles
//        staged in double-buffered LDS by the Tensor Data Mover
//        (tensor_load_to_lds + s_wait_tensorcnt), compute overlaps DMA.
// ---------------------------------------------------------------------------

typedef __attribute__((ext_vector_type(16))) _Float16 v16h;
typedef __attribute__((ext_vector_type(8)))  _Float16 v8h;
typedef __attribute__((ext_vector_type(8)))  float    v8f;
typedef __attribute__((ext_vector_type(4)))  unsigned int u32x4;
typedef __attribute__((ext_vector_type(8)))  int      i32x8;
typedef __attribute__((ext_vector_type(4)))  int      i32x4;

#define S_LEN 2048
#define E_DIM 1024
#define HEADS 16
#define DHEAD 64

#define BM 128
#define BN 128
#define BK 32
#define LDT 40   // padded row stride in halves -> 80B, multiple of 16B

#if defined(__HIP_DEVICE_COMPILE__) && __has_builtin(__builtin_amdgcn_tensor_load_to_lds)
#define HAVE_TDM 1
#else
#define HAVE_TDM 0
#endif

__device__ __forceinline__ v8f wmma_f16(v16h a, v16h b, v8f c) {
  return __builtin_amdgcn_wmma_f32_16x16x32_f16(false, a, false, b,
                                                (short)0, c, false, false);
}

// A-fragment (16x32, MxK): lane row = lane&15; elements 0..7 <-> k=hi*8+0..7,
// elements 8..15 <-> k=16+hi*8+0..7 (ISA 7.12.2).  Two 16B loads.
__device__ __forceinline__ v16h ld_afrag(const _Float16* rowp, int hi) {
  v8h lo = *(const v8h*)(rowp + hi * 8);
  v8h hh = *(const v8h*)(rowp + 16 + hi * 8);
  return __builtin_shufflevector(lo, hh, 0, 1, 2, 3, 4, 5, 6, 7, 8, 9, 10, 11,
                                 12, 13, 14, 15);
}

// B-fragment (32x16, KxN): lane col = lane&15; elements 0..7 <-> k=hi*8+0..7,
// elements 8..15 <-> k=16+hi*8+0..7 (matches A-fragment).
// colp = start of this column's contiguous K run.  Two 16B loads.
__device__ __forceinline__ v16h ld_bfrag(const _Float16* colp, int hi) {
  v8h lo = *(const v8h*)(colp + hi * 8);
  v8h hh = *(const v8h*)(colp + 16 + hi * 8);
  return __builtin_shufflevector(lo, hh, 0, 1, 2, 3, 4, 5, 6, 7, 8, 9, 10, 11,
                                 12, 13, 14, 15);
}

__device__ __forceinline__ void wait_tensorcnt0() {
#if __has_builtin(__builtin_amdgcn_s_wait_tensorcnt)
  __builtin_amdgcn_s_wait_tensorcnt(0);
#else
  asm volatile("s_wait_tensorcnt 0x0" ::: "memory");
#endif
}

#if HAVE_TDM
// Issue a TDM 2D tile load: tile_d1 rows x tile_d0 elements (f16), row
// stride d0_stride elements, into dense LDS at lds_byte_addr.
// D# packing per CDNA5 ISA.
// This toolchain's builtin is the 6-arg form:
//   (u32x4 g0, i32x8 g1, i32x4 g2, i32x4 g3, i32x8 pad, i32 cpol)
__device__ __forceinline__ void tdm_load_2d(unsigned int lds_byte_addr,
                                            const void* gptr,
                                            unsigned int tensor_d0,
                                            unsigned int tensor_d1,
                                            unsigned long long d0_stride,
                                            unsigned int tile_d0,
                                            unsigned int tile_d1) {
  unsigned long long ga = (unsigned long long)(size_t)gptr;
  u32x4 g0;
  g0.x = 1u;                                   // count=1 (valid user D#)
  g0.y = lds_byte_addr;                        // D#.lds_addr
  g0.z = (unsigned int)ga;                     // global_addr[31:0]
  g0.w = (unsigned int)((ga >> 32) & 0x01FFFFFFu) | (2u << 30);  // type=2
  i32x8 g1 = {};
  g1[0] = 1 << 16;                             // data_size=1 (2 bytes)
  g1[1] = (int)((tensor_d0 & 0xFFFFu) << 16);  // tensor_dim0[15:0]
  g1[2] = (int)((tensor_d0 >> 16) | ((tensor_d1 & 0xFFFFu) << 16));
  g1[3] = (int)((tensor_d1 >> 16) | (tile_d0 << 16));   // tile_dim0
  g1[4] = (int)tile_d1;                        // tile_dim1 (tile_dim2=0)
  g1[5] = (int)(unsigned int)(d0_stride & 0xFFFFFFFFull);
  g1[6] = (int)(unsigned int)((d0_stride >> 32) & 0xFFFFull);
  g1[7] = 0;
  i32x4 z4 = {};
  i32x8 z8 = {};
  __builtin_amdgcn_tensor_load_to_lds(g0, g1, z4, z4, z8, 0);
}
#endif

// ---------------------------------------------------------------------------
// Kernel 1: fused Q/K/V projection (x @ W + b), f16 outputs.
// Q,K stored [b,h,s,d]; V stored transposed [b,h,d,s] for the P@V B-frags.
// ---------------------------------------------------------------------------
__global__ __launch_bounds__(256) void qkv_gemm(
    const float* __restrict__ x,
    const float* __restrict__ Wq, const float* __restrict__ bq,
    const float* __restrict__ Wk, const float* __restrict__ bk,
    const float* __restrict__ Wv, const float* __restrict__ bv,
    _Float16* __restrict__ Qw, _Float16* __restrict__ Kw,
    _Float16* __restrict__ Vw) {
  __shared__ __align__(16) _Float16 As[2][BM][LDT];
  __shared__ __align__(16) _Float16 Bt[2][BN][LDT];  // transposed: [col][k]

  const int which = blockIdx.z;
  const float* W   = (which == 0) ? Wq : ((which == 1) ? Wk : Wv);
  const float* bia = (which == 0) ? bq : ((which == 1) ? bk : bv);
  _Float16*    dst = (which == 0) ? Qw : ((which == 1) ? Kw : Vw);

  const int N0   = blockIdx.x * BN;
  const int M0   = blockIdx.y * BM;
  const int tid  = threadIdx.x;
  const int lane = tid & 31;
  const int wave = tid >> 5;
  const int wm   = wave >> 1;  // 0..3: 32-row strip
  const int wn   = wave & 1;   // 0..1: 64-col strip
  const int n    = lane & 15;
  const int hi   = lane >> 4;

  float4 ra[4], rb[4];  // next-tile staging registers

  auto g_load = [&](int k0) {
#pragma unroll
    for (int j = 0; j < 4; ++j) {
      int idx = tid + 256 * j;
      ra[j] = *(const float4*)(x + (size_t)(M0 + (idx >> 3)) * E_DIM + k0 +
                               ((idx & 7) << 2));
      rb[j] = *(const float4*)(W + (size_t)(k0 + (idx >> 5)) * E_DIM + N0 +
                               ((idx & 31) << 2));
    }
  };
  auto l_store = [&](int buf) {
#pragma unroll
    for (int j = 0; j < 4; ++j) {
      int idx = tid + 256 * j;
      {
        int row = idx >> 3, c = (idx & 7) << 2;
        As[buf][row][c + 0] = (_Float16)ra[j].x;
        As[buf][row][c + 1] = (_Float16)ra[j].y;
        As[buf][row][c + 2] = (_Float16)ra[j].z;
        As[buf][row][c + 3] = (_Float16)ra[j].w;
      }
      {
        int k = idx >> 5, c = (idx & 31) << 2;  // transpose into Bt[col][k]
        Bt[buf][c + 0][k] = (_Float16)rb[j].x;
        Bt[buf][c + 1][k] = (_Float16)rb[j].y;
        Bt[buf][c + 2][k] = (_Float16)rb[j].z;
        Bt[buf][c + 3][k] = (_Float16)rb[j].w;
      }
    }
  };

  v8f acc[2][4] = {};

  g_load(0);
  l_store(0);
  __syncthreads();

  for (int k0 = 0; k0 < E_DIM; k0 += BK) {
    const int buf = (k0 >> 5) & 1;
    const bool nxt = (k0 + BK) < E_DIM;
    if (nxt) g_load(k0 + BK);  // next-tile HBM loads issued under the WMMAs

    v16h a[2], b[4];
#pragma unroll
    for (int mi = 0; mi < 2; ++mi)
      a[mi] = ld_afrag(&As[buf][wm * 32 + mi * 16 + n][0], hi);
#pragma unroll
    for (int ni = 0; ni < 4; ++ni)
      b[ni] = ld_bfrag(&Bt[buf][wn * 64 + ni * 16 + n][0], hi);
#pragma unroll
    for (int mi = 0; mi < 2; ++mi)
#pragma unroll
      for (int ni = 0; ni < 4; ++ni)
        acc[mi][ni] = wmma_f16(a[mi], b[ni], acc[mi][ni]);

    if (nxt) l_store(buf ^ 1);
    __syncthreads();
  }

#pragma unroll
  for (int mi = 0; mi < 2; ++mi)
#pragma unroll
    for (int ni = 0; ni < 4; ++ni) {
      int gn = N0 + wn * 64 + ni * 16 + n;
      float bv_ = bia[gn];
      int h = gn >> 6, d = gn & 63;
#pragma unroll
      for (int r = 0; r < 8; ++r) {
        int gm   = M0 + wm * 32 + mi * 16 + r + 8 * hi;
        int bb   = gm >> 11;          // / S_LEN
        int srow = gm & (S_LEN - 1);
        size_t off;
        if (which == 2)  // V^T: [b,h][d][s]
          off = ((size_t)(bb * HEADS + h) * DHEAD + d) * S_LEN + srow;
        else             // Q,K: [b,h][s][d]
          off = ((size_t)(bb * HEADS + h) * S_LEN + srow) * DHEAD + d;
        dst[off] = (_Float16)(acc[mi][ni][r] + bv_);
      }
    }
}

// ---------------------------------------------------------------------------
// Kernel 2: causal flash attention.  Block = 4 waves = 64 q rows of one
// (b,h).  64-key K/V tiles staged in double-buffered LDS by the TDM (wave 0
// issues tensor_load_to_lds for the next tile while all waves compute on the
// current one).  All waves iterate to the block causal limit; fully-masked
// subtiles are numerically inert in the online softmax.
// ---------------------------------------------------------------------------
__global__ __launch_bounds__(128) void attn_kernel(
    const _Float16* __restrict__ Qw, const _Float16* __restrict__ Kw,
    const _Float16* __restrict__ Vt, _Float16* __restrict__ Aout) {
  __shared__ __align__(16) _Float16 Ks[2][64][DHEAD];  // [key][d]
  __shared__ __align__(16) _Float16 Vs[2][64][64];     // [d][key]
  __shared__ __align__(16) _Float16 pbuf[4][16][LDT];  // per-wave P transpose

  const int lane = threadIdx.x & 31;
  const int wave = threadIdx.x >> 5;
  const int n    = lane & 15;
  const int hi   = lane >> 4;
  const int bh   = blockIdx.y;
  const int b    = bh >> 4;
  const int h    = bh & 15;
  const size_t base = (size_t)bh * S_LEN * DHEAD;  // extent of K and V^T slab
  const int Q0   = blockIdx.x * 64;
  const int q0   = Q0 + wave * 16;
  const int kmax = Q0 + 64;  // block-uniform causal limit (exclusive)

  // Q fragments (A layout), d split into two K=32 halves; vector loads.
  v16h qf[2];
#pragma unroll
  for (int p = 0; p < 2; ++p)
    qf[p] = ld_afrag(Qw + base + (size_t)(q0 + n) * DHEAD + p * 32, hi);

#if !HAVE_TDM
  v8h rk[4], rv[4];  // fallback staging registers
#endif

  auto stage_issue = [&](int buf, int k0) {
#if HAVE_TDM
    if (wave == 0) {
      // K tile: 64 rows (keys) x 64 d, row stride DHEAD.
      tdm_load_2d((unsigned int)(size_t)&Ks[buf][0][0],
                  Kw + base + (size_t)k0 * DHEAD, DHEAD, 64, DHEAD, DHEAD, 64);
      // V^T tile: 64 rows (d) x 64 keys, row stride S_LEN.
      tdm_load_2d((unsigned int)(size_t)&Vs[buf][0][0], Vt + base + k0, 64,
                  DHEAD, S_LEN, 64, DHEAD);
    }
#else
#pragma unroll
    for (int j = 0; j < 4; ++j) {
      int idx = (int)threadIdx.x + 128 * j;  // 0..511
      int row = idx >> 3, c = (idx & 7) << 3;
      rk[j] = *(const v8h*)(Kw + base + (size_t)(k0 + row) * DHEAD + c);
      rv[j] = *(const v8h*)(Vt + base + (size_t)row * S_LEN + k0 + c);
    }
#endif
  };
  auto stage_commit = [&](int buf) {
#if HAVE_TDM
    if (wave == 0) wait_tensorcnt0();
#else
#pragma unroll
    for (int j = 0; j < 4; ++j) {
      int idx = (int)threadIdx.x + 128 * j;
      int row = idx >> 3, c = (idx & 7) << 3;
      *(v8h*)&Ks[buf][row][c] = rk[j];
      *(v8h*)&Vs[buf][row][c] = rv[j];
    }
#endif
  };

  v8f   o[4] = {};
  float mrow[8], lrow[8];
#pragma unroll
  for (int r = 0; r < 8; ++r) { mrow[r] = -1e30f; lrow[r] = 0.0f; }

  stage_issue(0, 0);
  stage_commit(0);
  __syncthreads();

  for (int k0 = 0; k0 < kmax; k0 += 64) {
    const int buf = (k0 >> 6) & 1;
    const bool nxt = (k0 + 64) < kmax;
    if (nxt) stage_issue(buf ^ 1, k0 + 64);  // TDM DMA overlaps the WMMAs

#pragma unroll
    for (int kb = 0; kb < 64; kb += 32) {
      // ---- S = (Q K^T)/8, causal-masked -------------------------------
      v8f s[2];
#pragma unroll
      for (int ns = 0; ns < 2; ++ns) {
        v8f c = {};
        int kl = kb + ns * 16 + n;  // key within tile (this lane's column)
#pragma unroll
        for (int p = 0; p < 2; ++p) {
          v16h kf = ld_bfrag(&Ks[buf][kl][p * 32], hi);
          c = wmma_f16(qf[p], kf, c);
        }
        int key = k0 + kl;
#pragma unroll
        for (int r = 0; r < 8; ++r) {
          int q = q0 + r + 8 * hi;
          s[ns][r] = (key > q) ? -1e30f : c[r] * 0.125f;  // 1/sqrt(64)
        }
      }
      // ---- online softmax (row reductions within the 16-lane half) ---
      float alpha[8];
#pragma unroll
      for (int r = 0; r < 8; ++r) {
        float t = fmaxf(s[0][r], s[1][r]);
        t = fmaxf(t, __shfl_xor(t, 1, 32));
        t = fmaxf(t, __shfl_xor(t, 2, 32));
        t = fmaxf(t, __shfl_xor(t, 4, 32));
        t = fmaxf(t, __shfl_xor(t, 8, 32));
        float mnew = fmaxf(mrow[r], t);
        alpha[r]   = __expf(mrow[r] - mnew);
        float p0   = __expf(s[0][r] - mnew);
        float p1   = __expf(s[1][r] - mnew);
        float rs   = p0 + p1;
        rs += __shfl_xor(rs, 1, 32);
        rs += __shfl_xor(rs, 2, 32);
        rs += __shfl_xor(rs, 4, 32);
        rs += __shfl_xor(rs, 8, 32);
        lrow[r] = lrow[r] * alpha[r] + rs;
        mrow[r] = mnew;
        pbuf[wave][r + 8 * hi][n]      = (_Float16)p0;
        pbuf[wave][r + 8 * hi][16 + n] = (_Float16)p1;
#pragma unroll
        for (int t4 = 0; t4 < 4; ++t4) o[t4][r] *= alpha[r];
      }
      asm volatile("s_wait_dscnt 0x0" ::: "memory");  // wave-local RAW fence
      v16h pf = ld_afrag(&pbuf[wave][n][0], hi);      // P in A-frag layout
      // ---- O += P @ V -------------------------------------------------
#pragma unroll
      for (int t4 = 0; t4 < 4; ++t4) {
        v16h vf = ld_bfrag(&Vs[buf][t4 * 16 + n][kb], hi);
        o[t4] = wmma_f16(pf, vf, o[t4]);
      }
    }

    if (nxt) stage_commit(buf ^ 1);
    __syncthreads();
  }

  // Epilogue: normalize, write f16 attn output [b*S, E] row-major.
  float inv[8];
#pragma unroll
  for (int r = 0; r < 8; ++r) inv[r] = 1.0f / lrow[r];
#pragma unroll
  for (int t4 = 0; t4 < 4; ++t4) {
    int col = h * 64 + t4 * 16 + n;
#pragma unroll
    for (int r = 0; r < 8; ++r) {
      int q = q0 + r + 8 * hi;
      Aout[(size_t)(b * S_LEN + q) * E_DIM + col] =
          (_Float16)(o[t4][r] * inv[r]);
    }
  }
}

// ---------------------------------------------------------------------------
// Kernel 3: output projection.  out = attn_f16 @ Wo + bo, f32 result.
// ---------------------------------------------------------------------------
__global__ __launch_bounds__(256) void out_gemm(
    const _Float16* __restrict__ Af, const float* __restrict__ Wo,
    const float* __restrict__ bo, float* __restrict__ out) {
  __shared__ __align__(16) _Float16 As[2][BM][LDT];
  __shared__ __align__(16) _Float16 Bt[2][BN][LDT];

  const int N0   = blockIdx.x * BN;
  const int M0   = blockIdx.y * BM;
  const int tid  = threadIdx.x;
  const int lane = tid & 31;
  const int wave = tid >> 5;
  const int wm   = wave >> 1;
  const int wn   = wave & 1;
  const int n    = lane & 15;
  const int hi   = lane >> 4;

  v8h    raf[2];
  float4 rb[4];

  auto g_load = [&](int k0) {
#pragma unroll
    for (int j = 0; j < 2; ++j) {
      int idx = tid + 256 * j;
      raf[j] = *(const v8h*)(Af + (size_t)(M0 + (idx >> 2)) * E_DIM + k0 +
                             ((idx & 3) << 3));
    }
#pragma unroll
    for (int j = 0; j < 4; ++j) {
      int idx = tid + 256 * j;
      rb[j] = *(const float4*)(Wo + (size_t)(k0 + (idx >> 5)) * E_DIM + N0 +
                               ((idx & 31) << 2));
    }
  };
  auto l_store = [&](int buf) {
#pragma unroll
    for (int j = 0; j < 2; ++j) {
      int idx = tid + 256 * j;
      *(v8h*)&As[buf][idx >> 2][(idx & 3) << 3] = raf[j];
    }
#pragma unroll
    for (int j = 0; j < 4; ++j) {
      int idx = tid + 256 * j;
      int k = idx >> 5, c = (idx & 31) << 2;
      Bt[buf][c + 0][k] = (_Float16)rb[j].x;
      Bt[buf][c + 1][k] = (_Float16)rb[j].y;
      Bt[buf][c + 2][k] = (_Float16)rb[j].z;
      Bt[buf][c + 3][k] = (_Float16)rb[j].w;
    }
  };

  v8f acc[2][4] = {};

  g_load(0);
  l_store(0);
  __syncthreads();

  for (int k0 = 0; k0 < E_DIM; k0 += BK) {
    const int buf = (k0 >> 5) & 1;
    const bool nxt = (k0 + BK) < E_DIM;
    if (nxt) g_load(k0 + BK);

    v16h a[2], b[4];
#pragma unroll
    for (int mi = 0; mi < 2; ++mi)
      a[mi] = ld_afrag(&As[buf][wm * 32 + mi * 16 + n][0], hi);
#pragma unroll
    for (int ni = 0; ni < 4; ++ni)
      b[ni] = ld_bfrag(&Bt[buf][wn * 64 + ni * 16 + n][0], hi);
#pragma unroll
    for (int mi = 0; mi < 2; ++mi)
#pragma unroll
      for (int ni = 0; ni < 4; ++ni)
        acc[mi][ni] = wmma_f16(a[mi], b[ni], acc[mi][ni]);

    if (nxt) l_store(buf ^ 1);
    __syncthreads();
  }

#pragma unroll
  for (int mi = 0; mi < 2; ++mi)
#pragma unroll
    for (int ni = 0; ni < 4; ++ni) {
      int gn = N0 + wn * 64 + ni * 16 + n;
      float bv_ = bo[gn];
#pragma unroll
      for (int r = 0; r < 8; ++r) {
        int gm = M0 + wm * 32 + mi * 16 + r + 8 * hi;
        out[(size_t)gm * E_DIM + gn] = acc[mi][ni][r] + bv_;
      }
    }
}

// ---------------------------------------------------------------------------
extern "C" void kernel_launch(void* const* d_in, const int* in_sizes, int n_in,
                              void* d_out, int out_size, void* d_ws,
                              size_t ws_size, hipStream_t stream) {
  const float* x  = (const float*)d_in[0];
  const float* Wq = (const float*)d_in[1];
  const float* bq = (const float*)d_in[2];
  const float* Wk = (const float*)d_in[3];
  const float* bk = (const float*)d_in[4];
  const float* Wv = (const float*)d_in[5];
  const float* bv = (const float*)d_in[6];
  const float* Wo = (const float*)d_in[7];
  const float* bo = (const float*)d_in[8];
  float* out = (float*)d_out;

  const size_t per = (size_t)2 * HEADS * S_LEN * DHEAD;  // 4,194,304 halves
  _Float16* Qw = (_Float16*)d_ws;
  _Float16* Kw = Qw + per;
  _Float16* Vt = Kw + per;   // V stored transposed [b,h][d][s]
  _Float16* Aw = Vt + per;   // attn output [4096][1024]; total 32 MB

  dim3 g1(E_DIM / BN, (2 * S_LEN) / BM, 3);
  qkv_gemm<<<g1, dim3(256), 0, stream>>>(x, Wq, bq, Wk, bk, Wv, bv, Qw, Kw, Vt);

  dim3 g2(S_LEN / 64, 2 * HEADS);
  attn_kernel<<<g2, dim3(128), 0, stream>>>(Qw, Kw, Vt, Aw);

  dim3 g3(E_DIM / BN, (2 * S_LEN) / BM);
  out_gemm<<<g3, dim3(256), 0, stream>>>(Aw, Wo, bo, out);
}
